// VQVAE_251_66391604461612
// MI455X (gfx1250) — matrix operand order repaired and checked by reading, not compiled, not run on hardware
//
#include <hip/hip_runtime.h>

typedef __bf16 bf16;
typedef __bf16 v16bf __attribute__((ext_vector_type(16)));
typedef __bf16 v8bf  __attribute__((ext_vector_type(8)));
typedef float  v8f   __attribute__((ext_vector_type(8)));

#define BM 64
#define BN 64
#define BK 32
#define CTHREADS 256
#define LDSPAD 8   // row stride BK+8=40 elems: 16B alignment at cols 0/8/16/24

// ---------------------------------------------------------------------------
// Weight pre-conversion: f32 (OC, IC, KW) -> bf16 padded (OCp, KW*ICp),
// K reordered kw-major: o[oc][kw*ICp + ic] = w[oc][ic][kw], zero-filled pad.
// kw-major K means every BK-chunk of K has ONE kw and CONTIGUOUS ic.
// ---------------------------------------------------------------------------
__global__ void k_cvt_w(const float* __restrict__ w, bf16* __restrict__ o,
                        int OC, int IC, int KW, int ICp, size_t tot)
{
    size_t i = (size_t)blockIdx.x * blockDim.x + threadIdx.x;
    if (i >= tot) return;
    int Kp  = KW * ICp;
    int col = (int)(i % Kp);
    int row = (int)(i / Kp);
    int kw  = col / ICp;
    int ic  = col - kw * ICp;
    float v = (row < OC && ic < IC) ? w[((size_t)row * IC + ic) * KW + kw] : 0.f;
    o[i] = (bf16)v;
}

// ---------------------------------------------------------------------------
// 1D conv as implicit GEMM on the bf16 WMMA pipe, software-pipelined with
// double-buffered LDS. 8 wave32/block; wave = one 16-row M subtile x two
// 16-col time subtiles (one A fragment feeds two WMMAs).
// ---------------------------------------------------------------------------
template <int KW>
__global__ __launch_bounds__(CTHREADS)
void k_conv1d_wmma(const bf16* __restrict__ in, const bf16* __restrict__ Wb,
                   const float* __restrict__ bias, const bf16* __restrict__ skip,
                   bf16* __restrict__ out_bf, float* __restrict__ out_f,
                   int ICp, int OC, int Tin, int Tout,
                   int stride, int pad, int dil,
                   int relu_in, int relu_out)
{
    __shared__ __align__(16) unsigned short Asm[2][BM][BK + LDSPAD];
    __shared__ __align__(16) unsigned short Bsm[2][BN][BK + LDSPAD];

    const int tid   = threadIdx.x;
    const int lane  = tid & 31;
    const int wv    = tid >> 5;
    const int msub  = wv & 3;
    const int nhalf = wv >> 2;
    const int lmod  = lane & 15;
    const int lhalf = lane >> 4;

    const int n   = blockIdx.z;
    const int oc0 = blockIdx.y * BM;
    const int t0  = blockIdx.x * BN;

    const int Kp = KW * ICp;
    const int nk = Kp / BK;

    // A staging: one 16B load/store per thread, zero guards (padded weights)
    const int arow = tid >> 2;            // 64 rows
    const int aseg = (tid & 3) * 8;       // cols 0/8/16/24
    const size_t wrow = (size_t)(oc0 + arow) * Kp;

    // B staging: lane-consecutive t (coalesced); k = wave-id + e-group
    const int bk = tid >> 5;              // 0..7
    const int bt = tid & 31;              // 0..31
    const unsigned short* inw = (const unsigned short*)in;
    const unsigned short* pbase0 = inw + (size_t)n * ICp * Tin + (size_t)bk * Tin;
    const int tsb = (t0 + bt) * stride - pad;     // + kw*dil per chunk
    const int s32 = 32 * stride;
    const unsigned short relum = relu_in ? 0xFFFFu : 0u;

    uint4 areg;
    unsigned short breg[8];

    auto stage = [&](int kbase, int kw_, int ic0_) {
        areg = *(const uint4*)&Wb[wrow + kbase + aseg];
        const unsigned short* pb = pbase0 + (size_t)ic0_ * Tin;
        const int tb = tsb + kw_ * dil;
        #pragma unroll
        for (int e = 0; e < 8; ++e) {
            int ti = tb + (e >> 2) * s32;
            unsigned valid = (unsigned)ti < (unsigned)Tin;   // one unsigned cmp
            int tic = min(max(ti, 0), Tin - 1);
            unsigned short u = pb[(size_t)((e & 3) * 8) * Tin + tic];
            u &= (unsigned short)(0u - valid);                          // OOB -> 0
            u &= (unsigned short)~((unsigned short)((short)u >> 15) & relum); // ReLU
            breg[e] = u;
        }
    };

    v8f acc0 = {}, acc1 = {};
    int kw = 0, ic0 = 0;
    stage(0, 0, 0);

    for (int kc = 0; kc < nk; ++kc) {
        const int buf = kc & 1;
        // drain staged registers into this iteration's LDS buffer
        *(uint4*)&Asm[buf][arow][aseg] = areg;
        #pragma unroll
        for (int e = 0; e < 8; ++e)
            Bsm[buf][(e >> 2) * 32 + bt][(e & 3) * 8 + bk] = breg[e];
        __syncthreads();

        // issue next chunk's global loads; they overlap the WMMAs below
        ic0 += BK;
        if (ic0 >= ICp) { ic0 = 0; ++kw; }
        if (kc + 1 < nk) stage((kc + 1) * BK, kw, ic0);

        // fragments per CDNA5 16-bit WMMA layouts (b128 LDS loads)
        union U { v16bf v; v8bf h[2]; };
        U ua, ub0, ub1;
        const int ar = msub * 16 + lmod;
        ua.h[0] = *(const v8bf*)(const void*)&Asm[buf][ar][lhalf * 8];
        ua.h[1] = *(const v8bf*)(const void*)&Asm[buf][ar][16 + lhalf * 8];
        const int br0 = nhalf * 32 + lmod;
        ub0.h[0] = *(const v8bf*)(const void*)&Bsm[buf][br0][lhalf * 8];
        ub0.h[1] = *(const v8bf*)(const void*)&Bsm[buf][br0][16 + lhalf * 8];
        const int br1 = br0 + 16;
        ub1.h[0] = *(const v8bf*)(const void*)&Bsm[buf][br1][lhalf * 8];
        ub1.h[1] = *(const v8bf*)(const void*)&Bsm[buf][br1][16 + lhalf * 8];

        acc0 = __builtin_amdgcn_wmma_f32_16x16x32_bf16(false, ua.v, false, ub0.v,
                                                       (short)0, acc0, false, false);
        acc1 = __builtin_amdgcn_wmma_f32_16x16x32_bf16(false, ua.v, false, ub1.v,
                                                       (short)0, acc1, false, false);
    }

    // epilogue: C/D layout VGPR r -> M=r (lanes 0-15) / M=r+8 (lanes 16-31)
    #pragma unroll
    for (int j = 0; j < 2; ++j) {
        const v8f acc = j ? acc1 : acc0;
        const int t = t0 + nhalf * 32 + j * 16 + lmod;
        #pragma unroll
        for (int r = 0; r < 8; ++r) {
            int oc = oc0 + msub * 16 + r + lhalf * 8;
            if (oc < OC && t < Tout) {
                float v = acc[r] + bias[oc];
                size_t off = ((size_t)n * OC + oc) * Tout + t;
                if (skip)     v += (float)skip[off];
                if (relu_out) v = fmaxf(v, 0.f);
                if (out_f) out_f[off] = v;
                else       out_bf[off] = (bf16)v;
            }
        }
    }
}

// ---------------------------------------------------------------------------
// Layout helpers
// ---------------------------------------------------------------------------
__global__ void k_ntc_to_nct(const float* __restrict__ x, bf16* __restrict__ o,
                             int N, int T, int C, int Cp)  // -> (N,Cp,T), zero pad planes
{
    size_t i = (size_t)blockIdx.x * blockDim.x + threadIdx.x;
    size_t tot = (size_t)N * Cp * T;
    if (i >= tot) return;
    int t = (int)(i % T);
    size_t r = i / T;
    int c = (int)(r % Cp);
    int n = (int)(r / Cp);
    o[i] = (c < C) ? (bf16)x[((size_t)n * T + t) * C + c] : (bf16)0.f;
}

__global__ void k_nct_to_ntc(const float* __restrict__ in, float* __restrict__ o,
                             int N, int C, int T)  // (N,C,T) -> (N,T,C)
{
    size_t i = (size_t)blockIdx.x * blockDim.x + threadIdx.x;
    size_t tot = (size_t)N * T * C;
    if (i >= tot) return;
    int c = (int)(i % C);
    size_t r = i / C;
    int t = (int)(r % T);
    int n = (int)(r / T);
    o[i] = in[((size_t)n * C + c) * T + t];
}

__global__ void k_upsample2(const bf16* __restrict__ in, bf16* __restrict__ out,
                            int N, int C, int Tin)  // nearest 2x along T
{
    size_t i = (size_t)blockIdx.x * blockDim.x + threadIdx.x;
    int To = 2 * Tin;
    size_t tot = (size_t)N * C * To;
    if (i >= tot) return;
    int t = (int)(i % To);
    size_t r = i / To;                 // n*C + c
    out[i] = in[r * Tin + (t >> 1)];
}

// ---------------------------------------------------------------------------
// VQ: nearest codebook entry per (n,t) point; commit-loss sum + code counts.
// ---------------------------------------------------------------------------
#define CBT 256
__global__ __launch_bounds__(256)
void k_quantize(const float* __restrict__ h, const float* __restrict__ cb,
                bf16* __restrict__ dq, float* __restrict__ loss_acc,
                unsigned* __restrict__ counts, int N, int C, int T, int K)
{
    __shared__ float cs[CBT * 32];
    const int tid = threadIdx.x;
    const int p = blockIdx.x * blockDim.x + tid;   // N*T multiple of 256
    const int n = p / T, t = p % T;

    float xf[32];
    #pragma unroll
    for (int c = 0; c < 32; ++c)
        xf[c] = h[((size_t)n * C + c) * T + t];

    float best = 3.4e38f;
    int bi = 0;
    for (int c0 = 0; c0 < K; c0 += CBT) {
        __syncthreads();
        for (int j = tid; j < (CBT * 32) / 4; j += 256)
            ((float4*)cs)[j] = ((const float4*)(cb + (size_t)c0 * 32))[j];
        __syncthreads();
        for (int cc = 0; cc < CBT; ++cc) {
            const float* cv = &cs[cc * 32];
            float d = 0.f;
            #pragma unroll
            for (int c = 0; c < 32; ++c) {
                float df = xf[c] - cv[c];
                d = fmaf(df, df, d);
            }
            if (d < best) { best = d; bi = c0 + cc; }  // strict <: first-min = argmin
        }
    }
    #pragma unroll
    for (int c = 0; c < 32; ++c)
        dq[((size_t)n * C + c) * T + t] = (bf16)cb[(size_t)bi * 32 + c];
    atomicAdd(loss_acc, best);
    atomicAdd(&counts[bi], 1u);
}

__global__ void k_finalize(const float* __restrict__ loss_acc,
                           const unsigned* __restrict__ counts,
                           float* __restrict__ out2, int K, float invPC, float invP)
{
    __shared__ float red[256];
    const int tid = threadIdx.x;
    float s = 0.f;
    for (int k = tid; k < K; k += 256) {
        float pr = (float)counts[k] * invP;
        s += pr * logf(pr + 1e-7f);
    }
    red[tid] = s;
    __syncthreads();
    for (int w = 128; w > 0; w >>= 1) {
        if (tid < w) red[tid] += red[tid + w];
        __syncthreads();
    }
    if (tid == 0) {
        out2[0] = loss_acc[0] * invPC;   // commit loss
        out2[1] = expf(-red[0]);         // perplexity
    }
}

// ---------------------------------------------------------------------------
// Host-side orchestration
// ---------------------------------------------------------------------------
extern "C" void kernel_launch(void* const* d_in, const int* in_sizes, int n_in,
                              void* d_out, int out_size, void* d_ws, size_t ws_size,
                              hipStream_t stream)
{
    (void)in_sizes; (void)n_in; (void)out_size; (void)ws_size;

    int ii = 0;
    auto F = [&]() -> const float* { return (const float*)d_in[ii++]; };

    const float* x = F();
    const float *e_ci_w = F(), *e_ci_b = F();
    struct ResP { const float *w1, *b1, *w2, *b2; };
    struct DownP { const float *dw, *db; ResP r[3]; };
    DownP dn[2];
    for (int i = 0; i < 2; ++i) {
        dn[i].dw = F(); dn[i].db = F();
        for (int j = 0; j < 3; ++j) { dn[i].r[j].w1 = F(); dn[i].r[j].b1 = F();
                                      dn[i].r[j].w2 = F(); dn[i].r[j].b2 = F(); }
    }
    const float *e_co_w = F(), *e_co_b = F();
    const float *d_ci_w = F(), *d_ci_b = F();
    struct UpP { ResP r[3]; const float *cw, *cb; };
    UpP up[2];
    for (int i = 0; i < 2; ++i) {
        for (int j = 0; j < 3; ++j) { up[i].r[j].w1 = F(); up[i].r[j].b1 = F();
                                      up[i].r[j].w2 = F(); up[i].r[j].b2 = F(); }
        up[i].cw = F(); up[i].cb = F();
    }
    const float *cm_w = F(), *cm_b = F();
    const float *co_w = F(), *co_b = F();
    const float* codebook = F();

    // workspace carve-up
    const int N = 64, T0 = 512, CIN = 263, CINP = 288, WID = 512, CD = 32, K = 8192;
    char* p = (char*)d_ws;
    auto take = [&](size_t b) { char* r = p; p += (b + 255) & ~(size_t)255; return r; };
    bf16*  A0   = (bf16*)take((size_t)N * WID * T0 * 2);
    bf16*  A1   = (bf16*)take((size_t)N * WID * T0 * 2);
    char*  R    = take(38ull << 20);            // shared: A2 (bf16) / final f32 output
    bf16*  A2   = (bf16*)R;
    float* OUTF = (float*)R;
    bf16*  WREG = (bf16*)take(40ull << 20);     // padded, kw-major bf16 weights
    float* H    = (float*)take((size_t)N * CD * 128 * 4);
    bf16*  D0   = (bf16*)take((size_t)N * CD * 128 * 2);
    float* lossacc = (float*)take(256);
    unsigned* counts = (unsigned*)take((size_t)K * 4);

    hipMemsetAsync(lossacc, 0, 256, stream);
    hipMemsetAsync(counts, 0, (size_t)K * 4, stream);

    // ---- weight pre-conversion (f32 -> padded kw-major bf16), once per launch
    size_t woff = 0;
    auto cw = [&](const float* w, int OC, int IC, int KW) -> const bf16* {
        int OCp = (OC + 63) & ~63;
        int ICp = (IC + 31) & ~31;
        size_t tot = (size_t)OCp * KW * ICp;
        bf16* dst = WREG + woff;
        woff += (tot + 127) & ~(size_t)127;
        k_cvt_w<<<(unsigned)((tot + 255) / 256), 256, 0, stream>>>(w, dst, OC, IC, KW, ICp, tot);
        return dst;
    };
    const bf16* We_ci = cw(e_ci_w, WID, CIN, 3);
    struct ResW { const bf16 *w1, *w2; };
    struct DownW { const bf16* dw; ResW r[3]; };
    DownW dnw[2];
    for (int i = 0; i < 2; ++i) {
        dnw[i].dw = cw(dn[i].dw, WID, WID, 4);
        for (int j = 0; j < 3; ++j) {
            dnw[i].r[j].w1 = cw(dn[i].r[j].w1, WID, WID, 3);
            dnw[i].r[j].w2 = cw(dn[i].r[j].w2, WID, WID, 1);
        }
    }
    const bf16* We_co = cw(e_co_w, CD, WID, 3);
    const bf16* Wd_ci = cw(d_ci_w, WID, CD, 3);
    struct UpW { ResW r[3]; const bf16* cwp; };
    UpW upw[2];
    for (int i = 0; i < 2; ++i) {
        for (int j = 0; j < 3; ++j) {
            upw[i].r[j].w1 = cw(up[i].r[j].w1, WID, WID, 3);
            upw[i].r[j].w2 = cw(up[i].r[j].w2, WID, WID, 1);
        }
        upw[i].cwp = cw(up[i].cw, WID, WID, 3);
    }
    const bf16* Wcm = cw(cm_w, WID, WID, 3);
    const bf16* Wco = cw(co_w, CIN, WID, 3);

    auto conv = [&](const bf16* in, const bf16* Wb, const float* b, const bf16* sk,
                    bf16* obf, float* of, int ICp, int OC, int Tin, int Tout,
                    int KW, int st, int pad, int dil, int rin, int rout) {
        dim3 g((Tout + BN - 1) / BN, (OC + BM - 1) / BM, N);
        switch (KW) {
        case 1: k_conv1d_wmma<1><<<g, CTHREADS, 0, stream>>>(in, Wb, b, sk, obf, of,
                    ICp, OC, Tin, Tout, st, pad, dil, rin, rout); break;
        case 3: k_conv1d_wmma<3><<<g, CTHREADS, 0, stream>>>(in, Wb, b, sk, obf, of,
                    ICp, OC, Tin, Tout, st, pad, dil, rin, rout); break;
        case 4: k_conv1d_wmma<4><<<g, CTHREADS, 0, stream>>>(in, Wb, b, sk, obf, of,
                    ICp, OC, Tin, Tout, st, pad, dil, rin, rout); break;
        }
    };
    auto ups = [&](const bf16* in, bf16* out, int Tin) {
        size_t tot = (size_t)N * WID * 2 * Tin;
        k_upsample2<<<(unsigned)((tot + 255) / 256), 256, 0, stream>>>(in, out, N, WID, Tin);
    };

    // ======================= encoder =======================
    { size_t tot = (size_t)N * CINP * T0;
      k_ntc_to_nct<<<(unsigned)((tot + 255) / 256), 256, 0, stream>>>(x, A0, N, T0, CIN, CINP); }
    conv(A0, We_ci, e_ci_b, nullptr, A1, nullptr, CINP, WID, 512, 512, 3, 1, 1, 1, 0, 1);
    conv(A1, dnw[0].dw, dn[0].db, nullptr, A0, nullptr, WID, WID, 512, 256, 4, 2, 1, 1, 0, 0);
    conv(A0, dnw[0].r[0].w1, dn[0].r[0].b1, nullptr, A1, nullptr, WID, WID, 256, 256, 3, 1, 1, 1, 1, 0);
    conv(A1, dnw[0].r[0].w2, dn[0].r[0].b2, A0, A2, nullptr, WID, WID, 256, 256, 1, 1, 0, 1, 1, 0);
    conv(A2, dnw[0].r[1].w1, dn[0].r[1].b1, nullptr, A1, nullptr, WID, WID, 256, 256, 3, 1, 3, 3, 1, 0);
    conv(A1, dnw[0].r[1].w2, dn[0].r[1].b2, A2, A0, nullptr, WID, WID, 256, 256, 1, 1, 0, 1, 1, 0);
    conv(A0, dnw[0].r[2].w1, dn[0].r[2].b1, nullptr, A1, nullptr, WID, WID, 256, 256, 3, 1, 9, 9, 1, 0);
    conv(A1, dnw[0].r[2].w2, dn[0].r[2].b2, A0, A2, nullptr, WID, WID, 256, 256, 1, 1, 0, 1, 1, 0);
    conv(A2, dnw[1].dw, dn[1].db, nullptr, A0, nullptr, WID, WID, 256, 128, 4, 2, 1, 1, 0, 0);
    conv(A0, dnw[1].r[0].w1, dn[1].r[0].b1, nullptr, A1, nullptr, WID, WID, 128, 128, 3, 1, 1, 1, 1, 0);
    conv(A1, dnw[1].r[0].w2, dn[1].r[0].b2, A0, A2, nullptr, WID, WID, 128, 128, 1, 1, 0, 1, 1, 0);
    conv(A2, dnw[1].r[1].w1, dn[1].r[1].b1, nullptr, A1, nullptr, WID, WID, 128, 128, 3, 1, 3, 3, 1, 0);
    conv(A1, dnw[1].r[1].w2, dn[1].r[1].b2, A2, A0, nullptr, WID, WID, 128, 128, 1, 1, 0, 1, 1, 0);
    conv(A0, dnw[1].r[2].w1, dn[1].r[2].b1, nullptr, A1, nullptr, WID, WID, 128, 128, 3, 1, 9, 9, 1, 0);
    conv(A1, dnw[1].r[2].w2, dn[1].r[2].b2, A0, A2, nullptr, WID, WID, 128, 128, 1, 1, 0, 1, 1, 0);
    conv(A2, We_co, e_co_b, nullptr, nullptr, H, WID, CD, 128, 128, 3, 1, 1, 1, 0, 0);   // -> f32

    // ======================= quantizer =======================
    k_quantize<<<(N * 128) / 256, 256, 0, stream>>>(H, codebook, D0, lossacc, counts, N, CD, 128, K);

    // ======================= decoder =======================
    conv(D0, Wd_ci, d_ci_b, nullptr, A0, nullptr, CD, WID, 128, 128, 3, 1, 1, 1, 0, 1);
    conv(A0, upw[0].r[0].w1, up[0].r[0].b1, nullptr, A1, nullptr, WID, WID, 128, 128, 3, 1, 9, 9, 1, 0);
    conv(A1, upw[0].r[0].w2, up[0].r[0].b2, A0, A2, nullptr, WID, WID, 128, 128, 1, 1, 0, 1, 1, 0);
    conv(A2, upw[0].r[1].w1, up[0].r[1].b1, nullptr, A1, nullptr, WID, WID, 128, 128, 3, 1, 3, 3, 1, 0);
    conv(A1, upw[0].r[1].w2, up[0].r[1].b2, A2, A0, nullptr, WID, WID, 128, 128, 1, 1, 0, 1, 1, 0);
    conv(A0, upw[0].r[2].w1, up[0].r[2].b1, nullptr, A1, nullptr, WID, WID, 128, 128, 3, 1, 1, 1, 1, 0);
    conv(A1, upw[0].r[2].w2, up[0].r[2].b2, A0, A2, nullptr, WID, WID, 128, 128, 1, 1, 0, 1, 1, 0);
    ups(A2, A0, 128);
    conv(A0, upw[0].cwp, up[0].cb, nullptr, A1, nullptr, WID, WID, 256, 256, 3, 1, 1, 1, 0, 0);
    conv(A1, upw[1].r[0].w1, up[1].r[0].b1, nullptr, A2, nullptr, WID, WID, 256, 256, 3, 1, 9, 9, 1, 0);
    conv(A2, upw[1].r[0].w2, up[1].r[0].b2, A1, A0, nullptr, WID, WID, 256, 256, 1, 1, 0, 1, 1, 0);
    conv(A0, upw[1].r[1].w1, up[1].r[1].b1, nullptr, A2, nullptr, WID, WID, 256, 256, 3, 1, 3, 3, 1, 0);
    conv(A2, upw[1].r[1].w2, up[1].r[1].b2, A0, A1, nullptr, WID, WID, 256, 256, 1, 1, 0, 1, 1, 0);
    conv(A1, upw[1].r[2].w1, up[1].r[2].b1, nullptr, A2, nullptr, WID, WID, 256, 256, 3, 1, 1, 1, 1, 0);
    conv(A2, upw[1].r[2].w2, up[1].r[2].b2, A1, A0, nullptr, WID, WID, 256, 256, 1, 1, 0, 1, 1, 0);
    ups(A0, A1, 256);
    conv(A1, upw[1].cwp, up[1].cb, nullptr, A0, nullptr, WID, WID, 512, 512, 3, 1, 1, 1, 0, 0);
    conv(A0, Wcm, cm_b, nullptr, A1, nullptr, WID, WID, 512, 512, 3, 1, 1, 1, 0, 1);
    conv(A1, Wco, co_b, nullptr, nullptr, OUTF, WID, CIN, 512, 512, 3, 1, 1, 1, 0, 0);   // -> f32

    // ======================= outputs =======================
    { size_t tot = (size_t)N * T0 * CIN;
      k_nct_to_ntc<<<(unsigned)((tot + 255) / 256), 256, 0, stream>>>(OUTF, (float*)d_out, N, CIN, T0); }
    const float invPC = 1.f / (float)(N * 128 * CD);
    const float invP  = 1.f / (float)(N * 128);
    k_finalize<<<1, 256, 0, stream>>>(lossacc, counts,
                                      (float*)d_out + (size_t)N * T0 * CIN, K, invPC, invP);
}
